// DifferentialMultiheadAttention_16973710754147
// MI455X (gfx1250) — matrix operand order repaired and checked by reading, not compiled, not run on hardware
//
#include <hip/hip_runtime.h>
#include <hip/hip_bf16.h>
#include <stdint.h>

typedef _Float16 f16;
typedef __attribute__((ext_vector_type(16))) _Float16 v16h;
typedef __attribute__((ext_vector_type(8)))  float    v8f;
typedef __attribute__((ext_vector_type(4)))  uint32_t u32x4;
typedef __attribute__((ext_vector_type(4)))  int      i32x4;
typedef __attribute__((ext_vector_type(8)))  int      i32x8;

union U16x8 { v16h v; uint32_t u[8]; };
union F16U  { f16 h; uint16_t u; };

__device__ inline v8f zero8() {
  v8f z;
#pragma unroll
  for (int i = 0; i < 8; ++i) z[i] = 0.f;
  return z;
}

__device__ inline v8f wmma16(v16h a, v16h b, v8f c) {
  // D = A(16x32 f16) * B(32x16 f16) + C(16x16 f32)
  return __builtin_amdgcn_wmma_f32_16x16x32_f16(
      /*neg_a=*/false, a, /*neg_b=*/false, b,
      /*c_mod=*/(short)0, c, /*reuse_a=*/false, /*reuse_b=*/false);
}

// LDS byte offset of a __shared__ pointer: flat addresses in the LDS aperture
// keep the LDS byte address in the low 32 bits (ISA 10.2 aperture mapping).
__device__ inline uint32_t lds_off(const void* p) {
  return (uint32_t)(uintptr_t)p;
}

// ---------------------------------------------------------------------------
// Tensor Data Mover: 2-D f16 tile (tile_x elems per row, tile_rows rows) from
// global memory into LDS, with pad_amount=1 dword inserted every
// (8 << pad_interval) bytes so LDS rows land on our padded gather strides.
// D# bitfields per CDNA5 ISA §8.3/§8.4. Issued by one wave; completion via
// s_wait_tensorcnt; published to the workgroup by the following barrier.
// ---------------------------------------------------------------------------
__device__ inline void tdm_load_2d(uint32_t lds_byte_off, const void* gaddr,
                                   uint32_t tile_x, uint32_t tile_rows,
                                   uint32_t row_stride_elems,
                                   uint32_t pad_interval) {
  uint64_t ga = (uint64_t)(uintptr_t)gaddr;
  u32x4 g0;
  g0[0] = 1u;                                              // count=1, user D#
  g0[1] = lds_byte_off;                                    // lds_addr
  g0[2] = (uint32_t)(ga & 0xffffffffu);                    // global_addr[31:0]
  g0[3] = (uint32_t)((ga >> 32) & 0x1ffffffu) | (2u << 30); // [56:32] | type=2
  i32x8 g1;
  g1[0] = (int)((1u << 16) | (1u << 20) | (pad_interval << 22));
  //            data_size=2B  pad_enable   pad_interval; pad_amount=0 (1 dword)
  g1[1] = (int)(row_stride_elems << 16);  // tensor_dim0[15:0] (>= tile_x)
  g1[2] = (int)(8192u << 16);             // tensor_dim1[15:0] (large)
  g1[3] = (int)(tile_x << 16);            // tile_dim0
  g1[4] = (int)(tile_rows & 0xffffu);     // tile_dim1 (tile_dim2=0 -> 2-D)
  g1[5] = (int)row_stride_elems;          // tensor_dim0_stride[31:0]
  g1[6] = 0;
  g1[7] = 0;
  i32x4 z4 = {0, 0, 0, 0};
#if defined(__clang_major__) && (__clang_major__ >= 23)
  i32x8 z8 = {0, 0, 0, 0, 0, 0, 0, 0};
  __builtin_amdgcn_tensor_load_to_lds(g0, g1, z4, z4, z8, 0);
#else
  __builtin_amdgcn_tensor_load_to_lds(g0, g1, z4, z4, 0);
#endif
}

// Gather the 16-bit A-matrix (16x32) operand per CDNA5 ISA 7.12.2.
// base: LDS dword array laid out as [row][kpair]; stride in dwords.
__device__ inline v16h gather_a(const uint32_t* base, int stride) {
  const int lane = threadIdx.x & 31;
  const int hl = lane >> 4;
  const int row = lane & 15;
  U16x8 u;
#pragma unroll
  for (int v = 0; v < 8; ++v) {
    // half0: v0-3 -> K pairs 0-3 (K0..7), v4-7 -> pairs 8-11 (K16..23)
    // half1: v0-3 -> pairs 4-7 (K8..15),  v4-7 -> pairs 12-15 (K24..31)
    int p = (v & 3) + ((v >> 2) << 3) + (hl << 2);
    u.u[v] = base[row * stride + p];
  }
  return u.v;
}

// Gather the 16-bit B-matrix (32x16) operand per CDNA5 ISA 7.12.2.
// base: LDS dword array laid out as [col][kpair]; stride in dwords.
__device__ inline v16h gather_b(const uint32_t* base, int stride) {
  const int lane = threadIdx.x & 31;
  const int hl = lane >> 4;
  const int col = lane & 15;
  U16x8 u;
#pragma unroll
  for (int v = 0; v < 8; ++v) {
    // half0: v0-3 -> K pairs 0-3 (K0..7), v4-7 -> pairs 8-11 (K16..23)
    // half1: v0-3 -> pairs 4-7 (K8..15),  v4-7 -> pairs 12-15 (K24..31)
    int p = (v & 3) + ((v >> 2) << 3) + (hl << 2);
    u.u[v] = base[col * stride + p];
  }
  return u.v;
}

// ---------------------------------------------------------------------------
// Kernel 1: convert query (L,N,d) -> X (B*L, d) f16, and the 4 weights to f16
// ---------------------------------------------------------------------------
__global__ __launch_bounds__(256) void convert_kernel(
    const float* __restrict__ query,
    const float* __restrict__ Wq, const float* __restrict__ Wk,
    const float* __restrict__ Wv, const float* __restrict__ Wo,
    f16* __restrict__ Xh, f16* __restrict__ Wh) {
  size_t gid = (size_t)blockIdx.x * 256 + threadIdx.x;
  const size_t NX = (size_t)4096 * 1024;
  if (gid < NX) {
    int m = (int)(gid >> 10);          // m = b*1024 + l
    int k = (int)(gid & 1023);
    int b = m >> 10, l = m & 1023;
    Xh[gid] = (f16)query[((size_t)(l * 4 + b) << 10) + k];
  } else {
    size_t wid = gid - NX;             // 4 * 1M weight elements
    int w = (int)(wid >> 20);
    size_t idx = wid & ((1u << 20) - 1);
    const float* src = (w == 0) ? Wq : (w == 1) ? Wk : (w == 2) ? Wv : Wo;
    Wh[wid] = (f16)src[idx];
  }
}

// ---------------------------------------------------------------------------
// Kernel 2: lambda = exp(lq1.lk1) - exp(lq2.lk2) + 0.8
// ---------------------------------------------------------------------------
__global__ __launch_bounds__(32) void lambda_kernel(
    const float* __restrict__ lq1, const float* __restrict__ lk1,
    const float* __restrict__ lq2, const float* __restrict__ lk2,
    float* __restrict__ lam) {
  int l = threadIdx.x;  // 32 lanes, 64-dim dots
  float d1 = lq1[l] * lk1[l] + lq1[l + 32] * lk1[l + 32];
  float d2 = lq2[l] * lk2[l] + lq2[l + 32] * lk2[l + 32];
#pragma unroll
  for (int m = 16; m > 0; m >>= 1) {
    d1 += __shfl_xor(d1, m, 32);
    d2 += __shfl_xor(d2, m, 32);
  }
  if (l == 0) lam[0] = __expf(d1) - __expf(d2) + 0.8f;
}

// ---------------------------------------------------------------------------
// Kernel 3: WMMA GEMM  C[m,n] = (sum_k A[m,k]*B[k,n] + bias[n]) * scale
// A: f16 row-major MxK.  B: f16 row-major KxN.
// Block = 128 threads (4 waves); tile 64x128; each wave owns 32x64
// (8 WMMAs per 32-deep K chunk from 2 A-strips and 4 B-strips).
// A chunk is DMA'd by the Tensor Data Mover into the padded [row][kpair]
// LDS layout; B chunk is staged transposed by vector loads.
// out_mode 0: f16 row-major.  out_mode 1: f32, row m=b*1024+l written to
//             (l*4+b)*N + n  (the (L,N,d) output layout).
// ---------------------------------------------------------------------------
__global__ __launch_bounds__(128) void gemm_wmma(
    const f16* __restrict__ A, const f16* __restrict__ B,
    const float* __restrict__ bias, void* __restrict__ outp,
    int M, int K, int N, float scale, int out_mode) {
  __shared__ uint32_t As[64 * 17];     // [row][kpair], padded stride 17 (TDM)
  __shared__ uint32_t Bs32[128 * 17];  // [col][kpair] (B transposed)
  uint16_t* Bs = (uint16_t*)Bs32;

  const int tid = threadIdx.x;
  const int w = tid >> 5, lane = tid & 31;
  const int hl = lane >> 4, cidx = lane & 15;
  const int tileM = blockIdx.y * 64, tileN = blockIdx.x * 128;
  const int qm = (w >> 1) * 32, qn = (w & 1) * 64;
  const uint32_t* B32 = (const uint32_t*)B;
  const int N2 = N >> 1;

  v8f acc[2][4];
#pragma unroll
  for (int ti = 0; ti < 2; ++ti)
#pragma unroll
    for (int j = 0; j < 4; ++j) acc[ti][j] = zero8();

  for (int k0 = 0; k0 < K; k0 += 32) {
    __syncthreads();
    // A 64x32 chunk via TDM: rows land at stride 17 dwords (pad every 16).
    if (w == 0)
      tdm_load_2d(lds_off(As), A + (size_t)tileM * K + k0,
                  /*tile_x=*/32, /*rows=*/64, /*stride=*/(uint32_t)K,
                  /*pad_interval=*/3);
    // B 32x128 chunk staged transposed into [col][k]: 2048 dwords.
#pragma unroll
    for (int i = 0; i < 16; ++i) {
      int lin = tid + i * 128;
      int kk = lin >> 6, cp = lin & 63;  // cols 2cp, 2cp+1
      uint32_t v = B32[(size_t)(k0 + kk) * N2 + (tileN >> 1) + cp];
      Bs[(2 * cp) * 34 + kk] = (uint16_t)(v & 0xffffu);
      Bs[(2 * cp + 1) * 34 + kk] = (uint16_t)(v >> 16);
    }
    if (k0 + 32 < K)
      __builtin_prefetch(&B[(size_t)(k0 + 32 + (tid >> 2)) * N + tileN], 0, 1);
    if (w == 0) __builtin_amdgcn_s_wait_tensorcnt(0);
    __syncthreads();

    v16h a0 = gather_a(As + qm * 17, 17);
    v16h a1 = gather_a(As + (qm + 16) * 17, 17);
    v16h bt[4];
#pragma unroll
    for (int j = 0; j < 4; ++j) bt[j] = gather_b(Bs32 + (qn + j * 16) * 17, 17);
#pragma unroll
    for (int j = 0; j < 4; ++j) {
      acc[0][j] = wmma16(a0, bt[j], acc[0][j]);
      acc[1][j] = wmma16(a1, bt[j], acc[1][j]);
    }
  }

#pragma unroll
  for (int ti = 0; ti < 2; ++ti) {
#pragma unroll
    for (int j = 0; j < 4; ++j) {
      int ccol = tileN + qn + j * 16 + cidx;
      float bv = bias[ccol];
#pragma unroll
      for (int i = 0; i < 8; ++i) {
        int r = tileM + qm + ti * 16 + hl * 8 + i;  // C-layout row
        float vv = (acc[ti][j][i] + bv) * scale;
        if (out_mode == 0) {
          ((f16*)outp)[(size_t)r * N + ccol] = (f16)vv;
        } else {
          int bb = r >> 10, ll = r & 1023;
          ((float*)outp)[(size_t)(ll * 4 + bb) * N + ccol] = vv;
        }
      }
    }
  }
}

// ---------------------------------------------------------------------------
// Kernel 4: differential attention for one (batch, head-pair, 64-row q tile).
// Q pre-scaled by head_dim^-0.5. Two-pass softmax; O = P0/s0 V - lam P1/s1 V,
// then head-wise RMSNorm * g_norm * (1 - 0.8), f16 out row-major (B*L, 1024).
// Q and K tiles are DMA'd by the TDM into padded LDS; V is staged transposed.
// ---------------------------------------------------------------------------
__global__ __launch_bounds__(128) void attn_kernel(
    const f16* __restrict__ Qh, const f16* __restrict__ Kh,
    const f16* __restrict__ Vh, const float* __restrict__ gnorm,
    const float* __restrict__ lam_p, f16* __restrict__ AttO) {
  __shared__ uint32_t Qs[64 * 65];      // [qrow][dimpair], both sub-heads
  __shared__ uint32_t Ks[32 * 65];      // [key][dimpair], both sub-heads
  __shared__ uint32_t Vs32[128 * 17];   // [dim][keypair] (transposed V)
  __shared__ uint32_t Ps32[4][16 * 17]; // per-wave diff tile [row][keypair]
  uint16_t* Vs = (uint16_t*)Vs32;

  const int tid = threadIdx.x;
  const int w = tid >> 5, lane = tid & 31;
  const int hl = lane >> 4, cidx = lane & 15;
  const int qb = blockIdx.x, h = blockIdx.y, b = blockIdx.z;
  const float lam = lam_p[0];

  const uint32_t* V32 = (const uint32_t*)Vh;
  const size_t rowBase = (size_t)b * 1024 + (size_t)qb * 64;
  const size_t keyBase = (size_t)b * 1024;

  // Stage the 64x128(f16) Q tile (sub-heads 2h, 2h+1 adjacent cols) via TDM.
  if (w == 0) {
    tdm_load_2d(lds_off(Qs), Qh + rowBase * 1024 + (size_t)h * 128,
                /*tile_x=*/128, /*rows=*/64, /*stride=*/1024,
                /*pad_interval=*/5);  // pad 1 dword per 64 -> stride 65
    __builtin_amdgcn_s_wait_tensorcnt(0);
  }
  __syncthreads();

  v16h q[2][2];  // [sub-head e][k-chunk kc]
#pragma unroll
  for (int e = 0; e < 2; ++e)
#pragma unroll
    for (int kc = 0; kc < 2; ++kc)
      q[e][kc] = gather_a(Qs + (w * 16) * 65 + e * 32 + kc * 16, 65);

  float ms[2][8], ss[2][8];
#pragma unroll
  for (int e = 0; e < 2; ++e)
#pragma unroll
    for (int i = 0; i < 8; ++i) { ms[e][i] = -1e30f; ss[e][i] = 0.f; }

  // ---------- Pass 1: online row max & sum-exp per sub-head ----------
  for (int kb = 0; kb < 1024; kb += 32) {
    __syncthreads();
    if (w == 0) {
      tdm_load_2d(lds_off(Ks), Kh + (keyBase + kb) * 1024 + (size_t)h * 128,
                  128, 32, 1024, 5);
      __builtin_amdgcn_s_wait_tensorcnt(0);
    }
    __syncthreads();
#pragma unroll
    for (int e = 0; e < 2; ++e) {
#pragma unroll
      for (int j = 0; j < 2; ++j) {
        v8f acc = zero8();
        acc = wmma16(q[e][0], gather_b(Ks + (j * 16) * 65 + e * 32 + 0, 65), acc);
        acc = wmma16(q[e][1], gather_b(Ks + (j * 16) * 65 + e * 32 + 16, 65), acc);
#pragma unroll
        for (int i = 0; i < 8; ++i) {
          float v = acc[i];
          float rm = v;
          rm = fmaxf(rm, __shfl_xor(rm, 1, 32));
          rm = fmaxf(rm, __shfl_xor(rm, 2, 32));
          rm = fmaxf(rm, __shfl_xor(rm, 4, 32));
          rm = fmaxf(rm, __shfl_xor(rm, 8, 32));
          float mn = fmaxf(ms[e][i], rm);
          float pe = __expf(v - mn);
          pe += __shfl_xor(pe, 1, 32);
          pe += __shfl_xor(pe, 2, 32);
          pe += __shfl_xor(pe, 4, 32);
          pe += __shfl_xor(pe, 8, 32);
          ss[e][i] = ss[e][i] * __expf(ms[e][i] - mn) + pe;
          ms[e][i] = mn;
        }
      }
    }
  }

  float inv0[8], inv1[8];
#pragma unroll
  for (int i = 0; i < 8; ++i) {
    inv0[i] = 1.f / ss[0][i];
    inv1[i] = lam / ss[1][i];
  }

  v8f o[8];
#pragma unroll
  for (int t = 0; t < 8; ++t) o[t] = zero8();

  // ---------- Pass 2: O += (P0/s0 - lam*P1/s1) @ V ----------
  for (int kb = 0; kb < 1024; kb += 32) {
    __syncthreads();
    if (w == 0)
      tdm_load_2d(lds_off(Ks), Kh + (keyBase + kb) * 1024 + (size_t)h * 128,
                  128, 32, 1024, 5);
#pragma unroll
    for (int i = 0; i < 16; ++i) {
      int lin = tid + i * 128;
      int ky = lin >> 6, pp = lin & 63;
      uint32_t vv = V32[(keyBase + kb + ky) * 512 + h * 64 + pp];
      Vs[(2 * pp) * 34 + ky] = (uint16_t)(vv & 0xffffu);
      Vs[(2 * pp + 1) * 34 + ky] = (uint16_t)(vv >> 16);
    }
    if (w == 0) __builtin_amdgcn_s_wait_tensorcnt(0);
    __syncthreads();

    uint16_t* Ph = (uint16_t*)(&Ps32[w][0]);
#pragma unroll
    for (int j = 0; j < 2; ++j) {
      v8f a0 = zero8(), a1 = zero8();
      a0 = wmma16(q[0][0], gather_b(Ks + (j * 16) * 65 + 0, 65), a0);
      a0 = wmma16(q[0][1], gather_b(Ks + (j * 16) * 65 + 16, 65), a0);
      a1 = wmma16(q[1][0], gather_b(Ks + (j * 16) * 65 + 32, 65), a1);
      a1 = wmma16(q[1][1], gather_b(Ks + (j * 16) * 65 + 48, 65), a1);
#pragma unroll
      for (int i = 0; i < 8; ++i) {
        float d = __expf(a0[i] - ms[0][i]) * inv0[i] -
                  __expf(a1[i] - ms[1][i]) * inv1[i];
        F16U cv; cv.h = (f16)d;
        Ph[(i + hl * 8) * 34 + j * 16 + cidx] = cv.u;  // C-layout -> LDS
      }
    }
    __syncthreads();

    v16h pa = gather_a(&Ps32[w][0], 17);  // re-layout as A operand (16x32 keys)
#pragma unroll
    for (int t = 0; t < 8; ++t)
      o[t] = wmma16(pa, gather_b(Vs32 + (t * 16) * 17, 17), o[t]);
  }

  // ---------- Finalize: RMSNorm over 128 dims, * g_norm * (1 - 0.8) ----------
#pragma unroll
  for (int i = 0; i < 8; ++i) {
    float sq = 0.f;
#pragma unroll
    for (int t = 0; t < 8; ++t) sq += o[t][i] * o[t][i];
    sq += __shfl_xor(sq, 1, 32);
    sq += __shfl_xor(sq, 2, 32);
    sq += __shfl_xor(sq, 4, 32);
    sq += __shfl_xor(sq, 8, 32);
    float sc = rsqrtf(sq * (1.f / 128.f) + 1e-5f) * 0.2f;
    size_t row = rowBase + (size_t)(w * 16 + hl * 8 + i);
#pragma unroll
    for (int t = 0; t < 8; ++t) {
      float gv = gnorm[t * 16 + cidx];
      AttO[row * 1024 + h * 128 + t * 16 + cidx] = (f16)(o[t][i] * sc * gv);
    }
  }
}

// ---------------------------------------------------------------------------
extern "C" void kernel_launch(void* const* d_in, const int* in_sizes, int n_in,
                              void* d_out, int out_size, void* d_ws,
                              size_t ws_size, hipStream_t stream) {
  (void)in_sizes; (void)n_in; (void)out_size; (void)ws_size;
  const float* query = (const float*)d_in[0];
  const float* Wq = (const float*)d_in[3];
  const float* bq = (const float*)d_in[4];
  const float* Wk = (const float*)d_in[5];
  const float* bk = (const float*)d_in[6];
  const float* Wv = (const float*)d_in[7];
  const float* bv = (const float*)d_in[8];
  const float* Wo = (const float*)d_in[9];
  const float* bo = (const float*)d_in[10];
  const float* gn = (const float*)d_in[11];
  const float* lq1 = (const float*)d_in[12];
  const float* lk1 = (const float*)d_in[13];
  const float* lq2 = (const float*)d_in[14];
  const float* lk2 = (const float*)d_in[15];

  char* ws = (char*)d_ws;
  const size_t MB = 1024u * 1024u;
  f16* Xh  = (f16*)(ws + 0 * MB);   // 8 MB   (4096x1024 f16)
  f16* Wh  = (f16*)(ws + 8 * MB);   // 8 MB   (Wq,Wk,Wv,Wo f16, contiguous)
  f16* Wqh = Wh;
  f16* Wkh = Wh + (1u << 20);
  f16* Wvh = Wh + (2u << 20);
  f16* Woh = Wh + (3u << 20);
  f16* Qh  = (f16*)(ws + 16 * MB);  // 8 MB (pre-scaled by 0.125)
  f16* Kh  = (f16*)(ws + 24 * MB);  // 8 MB
  f16* Vh  = (f16*)(ws + 32 * MB);  // 8 MB
  f16* Ah  = (f16*)(ws + 40 * MB);  // 8 MB attention output (B*L, 1024)
  float* lam = (float*)(ws + 48 * MB);

  // 1) f32 -> f16 conversions (+ layout swap for X).
  convert_kernel<<<32768, 256, 0, stream>>>(query, Wq, Wk, Wv, Wo, Xh, Wh);
  // 2) lambda scalar.
  lambda_kernel<<<1, 32, 0, stream>>>(lq1, lk1, lq2, lk2, lam);

  // 3) QKV projections. Q pre-scaled by head_dim^-0.5 = 0.125.
  dim3 ggrid(8, 64);  // (N/128, M/64), M=4096, N=K=1024
  gemm_wmma<<<ggrid, 128, 0, stream>>>(Xh, Wqh, bq, Qh, 4096, 1024, 1024, 0.125f, 0);
  gemm_wmma<<<ggrid, 128, 0, stream>>>(Xh, Wkh, bk, Kh, 4096, 1024, 1024, 1.0f, 0);
  gemm_wmma<<<ggrid, 128, 0, stream>>>(Xh, Wvh, bv, Vh, 4096, 1024, 1024, 1.0f, 0);

  // 4) Differential attention: grid (q-tiles=16, h_eff=8, batch=4).
  attn_kernel<<<dim3(16, 8, 4), 128, 0, stream>>>(Qh, Kh, Vh, gn, lam, Ah);

  // 5) Output projection, f32 result with (B,L)->(L,N) remap.
  gemm_wmma<<<ggrid, 128, 0, stream>>>(Ah, Woh, bo, d_out, 4096, 1024, 1024, 1.0f, 1);
}
